// PointwiseConvolution_49022756716914
// MI455X (gfx1250) — matrix operand order repaired and checked by reading, not compiled, not run on hardware
//
#include <hip/hip_runtime.h>
#include <hip/hip_bf16.h>
#include <math.h>

// ---------------------------------------------------------------------------
// Problem constants (from reference)
// ---------------------------------------------------------------------------
#define N_NODES 32768
#define N_EDGES 262144
#define N_TYPES 10
#define EMB     16
#define NB      20
#define HID     64
#define OUTC    54            // 16 node_emb cols + 38 message cols
#define NMSG    38            // 16 (S) + 12 (V*3) + 10 (T*5)
#define SILU_NORM 1.679177f
#define KPAD    1032          // Bf row stride in halfs (2064B: 16B-aligned, 4-bank stagger)

typedef __attribute__((ext_vector_type(16))) _Float16 v16h;
typedef __attribute__((ext_vector_type(8)))  _Float16 v8h;
typedef __attribute__((ext_vector_type(2)))  _Float16 v2h;
typedef __attribute__((ext_vector_type(8)))  float    v8f;
typedef __attribute__((ext_vector_type(4)))  float    v4f;

union U16 {            // fragment assembly without elementwise cvt chains
  v16h v;
  v8h  p[2];
  v2h  q[8];
};

// v_wmma_f32_16x16x32_f16: D = A(16x32 f16) * B(32x16 f16) + C(16x16 f32)
#define WMMA(a, b, c) \
  __builtin_amdgcn_wmma_f32_16x16x32_f16(false, (a), false, (b), (short)0, (c), false, false)

// Intra-wave LDS producer->consumer fence (DS ops in-order per wave).
__device__ __forceinline__ void lds_fence_wave() {
  __asm__ __volatile__("s_wait_dscnt 0" ::: "memory");
}

// ---------------------------------------------------------------------------
// Kernel 0: zero the output (message columns are accumulated atomically)
// ---------------------------------------------------------------------------
__global__ void zero_kernel(float* __restrict__ out, int n) {
  int i = blockIdx.x * blockDim.x + threadIdx.x;
  if (i < n) out[i] = 0.0f;
}

// ---------------------------------------------------------------------------
// Kernel 1: node_emb = (x @ W1 / sqrt(10)) @ W2 / sqrt(64)  -> out[:, 0:16]
// One wave per 16-node tile; 4 waves / block. Weights staged transposed
// ([n][K], padded rows) so B fragments are contiguous 2x ds_load_b128.
// ---------------------------------------------------------------------------
__global__ __launch_bounds__(128) void node_kernel(
    const float* __restrict__ x, const float* __restrict__ W1,
    const float* __restrict__ W2, float* __restrict__ out) {
  __shared__ _Float16 W1sT[HID * 40];    // [n][K], K padded 10->40, *1/sqrt(10)
  __shared__ _Float16 W2sT[EMB * 72];    // [n][K], K padded 64->72, * 1/8
  __shared__ _Float16 tst[4][16 * HID];  // f16 staging of intermediate t (16x64)

  const float rs10 = 0.31622776601f;
  for (int idx = threadIdx.x; idx < HID * 40; idx += blockDim.x) {
    const int n = idx / 40, K = idx % 40;
    W1sT[idx] = (_Float16)((K < N_TYPES) ? W1[K * HID + n] * rs10 : 0.0f);
  }
  for (int idx = threadIdx.x; idx < EMB * 72; idx += blockDim.x) {
    const int n = idx / 72, K = idx % 72;
    W2sT[idx] = (_Float16)((K < HID) ? W2[K * EMB + n] * 0.125f : 0.0f);
  }
  __syncthreads();

  const int wave = threadIdx.x >> 5;
  const int lane = threadIdx.x & 31;
  const int l16  = lane & 15;
  const int hi   = lane >> 4;            // half-wave id (A/B fragment K split)
  const int tile = blockIdx.x * 4 + wave;
  const int row  = tile * 16 + l16;

  // Each half-wave redundantly loads its row of x (zero-padded K).
  float xr[32];
#pragma unroll
  for (int i = 0; i < 32; ++i) xr[i] = 0.0f;
#pragma unroll
  for (int i = 0; i < N_TYPES; ++i) xr[i] = x[row * N_TYPES + i];

  // A fragment: lane holds row M=l16; element j -> K=j+16*hi
  v16h af;
#pragma unroll
  for (int j = 0; j < 16; ++j) {
    const int K0 = j;
    af[j] = (_Float16)(hi ? xr[K0 + 16] : xr[K0]);
  }

  const v8f z8 = {0.f, 0.f, 0.f, 0.f, 0.f, 0.f, 0.f, 0.f};
  v8f acc[4] = {z8, z8, z8, z8};

  // t = x @ W1 * rs10 : B fragment = 16 contiguous halfs (K = 16*hi + j)
#pragma unroll
  for (int nt = 0; nt < 4; ++nt) {
    const _Float16* bp = &W1sT[(nt * 16 + l16) * 40 + hi * 16];
    U16 b;
    b.p[0] = *(const v8h*)(bp);
    b.p[1] = *(const v8h*)(bp + 8);
    acc[nt] = WMMA(af, b.v, acc[nt]);
  }

  // Stage t (f16) in LDS: D layout -> plain [row][col]
  _Float16* tw = tst[wave];
#pragma unroll
  for (int nt = 0; nt < 4; ++nt) {
#pragma unroll
    for (int r = 0; r < 8; ++r) {
      tw[(r + hi * 8) * HID + nt * 16 + l16] = (_Float16)acc[nt][r];
    }
  }
  lds_fence_wave();

  // node_emb = t @ W2 * 0.125   (K=64 -> 2 k-steps, 1 N-tile)
  const _Float16* myt = &tst[wave][l16 * HID];
  v8f d = z8;
#pragma unroll
  for (int ks = 0; ks < 2; ++ks) {
    // A: j<8 -> K=ks*32+8hi+j ; j>=8 -> K=ks*32+16+8hi+(j-8): two v8h chunks
    const _Float16* ap = myt + ks * 32 + hi * 8;
    U16 a;
    a.p[0] = *(const v8h*)(ap);
    a.p[1] = *(const v8h*)(ap + 16);
    const _Float16* bp = &W2sT[l16 * 72 + ks * 32 + hi * 8];
    U16 b;
    b.p[0] = *(const v8h*)(bp);
    b.p[1] = *(const v8h*)(bp + 16);
    d = WMMA(a.v, b.v, d);
  }

  // Write node_emb into out[:, 0:16] (also used as z-gather source by edges)
#pragma unroll
  for (int r = 0; r < 8; ++r) {
    const int nrow = tile * 16 + r + hi * 8;
    out[(size_t)nrow * OUTC + l16] = d[r];
  }
}

// ---------------------------------------------------------------------------
// Kernel 2: fused edge pipeline.
// One wave per 16-edge tile; 4 waves / block.
//   demb(16x20) --WMMA--> h(16x64) --silu--> hz outer-product fused GEMM:
//   D(16x22) = hz(16x1024) @ BfT(1024x22), BfT = reshaped W_tp2 transposed
//   [c][k] so each B fragment is 2x ds_load_b128. Then per-edge sh multiply
//   + atomic scatter-add into out[:, 16:54].
// ---------------------------------------------------------------------------
__global__ __launch_bounds__(128) void edge_kernel(
    const float* __restrict__ pos, const int* __restrict__ ei,
    const float* __restrict__ W_tp1, const float* __restrict__ W_tp2,
    float* __restrict__ out) {
  __shared__ _Float16 BfT[22 * KPAD];    // fused W_tp2, [c][kappa] (45.4 KB)
  __shared__ _Float16 W1sT[HID * 40];    // W_tp1 [n][K], K padded 20->40
  __shared__ char     scr[4][2048];      // per-wave: h stage (f16 16x64),
                                         // then reused as D stage (f32 16x24)

  // --- cooperative fill of fused B matrix: BfT[c][kappa], kappa = k*16+i ---
  // c<16: Ws col = i*16+c ; c<20: Wv col = 256+i*4+(c-16) ; else 320+i*2+(c-20)
  // scale = (1/sqrt(HID)) * (1/sqrt(EMB)) * (1/sqrt(num_neighbors=8))
  const float SCALE2 = 0.125f * 0.25f * 0.35355339059f;
  for (int idx = threadIdx.x; idx < 22 * 1024; idx += blockDim.x) {
    const int c = idx / 1024, kap = idx % 1024;
    const int k = kap >> 4, i = kap & 15;
    int col;
    if (c < 16)      col = i * 16 + c;
    else if (c < 20) col = 256 + i * 4 + (c - 16);
    else             col = 320 + i * 2 + (c - 20);
    BfT[c * KPAD + kap] = (_Float16)(W_tp2[k * 352 + col] * SCALE2);
  }
  const float rs20 = 0.22360679775f;  // 1/sqrt(20)
  for (int idx = threadIdx.x; idx < HID * 40; idx += blockDim.x) {
    const int n = idx / 40, K = idx % 40;
    W1sT[idx] = (_Float16)((K < NB) ? W_tp1[K * HID + n] * rs20 : 0.0f);
  }
  __syncthreads();

  const int wave = threadIdx.x >> 5;
  const int lane = threadIdx.x & 31;
  const int l16  = lane & 15;
  const int hi   = lane >> 4;
  const int tile = blockIdx.x * 4 + wave;
  const int e    = tile * 16 + l16;       // each half-wave covers all 16 edges

  // --- per-edge geometry (redundant across half-waves) ---------------------
  const int se = ei[e];
  const int de = ei[N_EDGES + e];
  const float vx = pos[de * 3 + 0] - pos[se * 3 + 0];
  const float vy = pos[de * 3 + 1] - pos[se * 3 + 1];
  const float vz = pos[de * 3 + 2] - pos[se * 3 + 2];
  const float dist = sqrtf(vx * vx + vy * vy + vz * vz + 1e-12f);
  const float ri = 1.0f / dist;
  const float ux = vx * ri, uy = vy * ri, uz = vz * ri;

  const float S3 = 1.73205080757f, S15 = 3.87298334621f, S5 = 2.2360679775f;
  float sh[9];
  sh[0] = 1.0f;
  sh[1] = S3 * ux; sh[2] = S3 * uy; sh[3] = S3 * uz;
  sh[4] = S15 * ux * uz;
  sh[5] = S15 * ux * uy;
  sh[6] = S5 * (uy * uy - 0.5f * (ux * ux + uz * uz));
  sh[7] = S15 * uy * uz;
  sh[8] = 0.5f * S15 * (uz * uz - ux * ux);

  // radial bump embedding (branchless, zero-padded to 32 for fragment build)
  float demb[32];
#pragma unroll
  for (int b = 0; b < 32; ++b) demb[b] = 0.0f;
  const float step = 5.0f / 21.0f;
#pragma unroll
  for (int b = 0; b < NB; ++b) {
    const float dd = (dist - (float)(b + 1) * step) / step;
    const bool m = fabsf(dd) < 1.0f;
    const float dsv = m ? dd : 0.0f;                       // cndmask
    const float f = 1.14136f *
        __expf(2.0f - 1.0f / (1.0f + dsv) - 1.0f / (1.0f - dsv)) *
        4.47213595f;                                       // * sqrt(NB)
    demb[b] = m ? f : 0.0f;                                // cndmask
  }

  // gather z = node_emb[src] from out[:, 0:16]
  float zv[16];
#pragma unroll
  for (int i = 0; i < 16; ++i) zv[i] = out[(size_t)se * OUTC + i];

  // --- GEMM 1: h = SILU_NORM * silu(demb @ W_tp1 / sqrt(NB)) ---------------
  v16h af;
#pragma unroll
  for (int j = 0; j < 16; ++j) {
    const int K0 = j;
    af[j] = (_Float16)(hi ? demb[K0 + 16] : demb[K0]);
  }
  const v8f z8 = {0.f, 0.f, 0.f, 0.f, 0.f, 0.f, 0.f, 0.f};
  v8f acc1[4] = {z8, z8, z8, z8};
#pragma unroll
  for (int nt = 0; nt < 4; ++nt) {
    const _Float16* bp = &W1sT[(nt * 16 + l16) * 40 + hi * 16];
    U16 b;
    b.p[0] = *(const v8h*)(bp);
    b.p[1] = *(const v8h*)(bp + 8);
    acc1[nt] = WMMA(af, b.v, acc1[nt]);
  }

  // silu + stage h (f16) per wave (scratch region, D-fragment scatter layout)
  _Float16* hw = (_Float16*)scr[wave];
#pragma unroll
  for (int nt = 0; nt < 4; ++nt) {
#pragma unroll
    for (int r = 0; r < 8; ++r) {
      const float v = acc1[nt][r];
      const float s = SILU_NORM * v / (1.0f + __expf(-v));
      hw[(r + hi * 8) * HID + nt * 16 + l16] = (_Float16)s;
    }
  }
  lds_fence_wave();

  // --- GEMM 2 (fused): D = (h (x) z)(16x1024) @ BfT(1024x22) ---------------
  // Pre-pack the 8 relevant z values (i = (j&7) + 8*hi) as f16 pairs.
  v2h zf2[4];
#pragma unroll
  for (int p = 0; p < 4; ++p) {
    const float za = hi ? zv[2 * p + 8] : zv[2 * p];
    const float zb = hi ? zv[2 * p + 9] : zv[2 * p + 1];
    v2h t = {(_Float16)za, (_Float16)zb};
    zf2[p] = t;
  }
  const _Float16* myh = (const _Float16*)scr[wave] + l16 * HID;
  const _Float16* bp0 = &BfT[l16 * KPAD];             // N-tile 0 column
  const int  n1   = 16 + l16;
  const bool n1ok = (n1 < 22);
  const _Float16* bp1 = &BfT[(n1ok ? n1 : 0) * KPAD]; // N-tile 1, clamped
  const v8h z8h = {(_Float16)0, (_Float16)0, (_Float16)0, (_Float16)0,
                   (_Float16)0, (_Float16)0, (_Float16)0, (_Float16)0};

  v8f acc2[2] = {z8, z8};
#pragma unroll 2
  for (int ks = 0; ks < 32; ++ks) {
    // h pair for this k-step (k = 2ks + (j>=8)), one ds_load_b32
    const v2h hp = *(const v2h*)(myh + 2 * ks);
    v2h h00 = {hp.x, hp.x};
    v2h h11 = {hp.y, hp.y};
    U16 a;
#pragma unroll
    for (int p = 0; p < 4; ++p) a.q[p] = h00 * zf2[p];       // v_pk_mul_f16
#pragma unroll
    for (int p = 0; p < 4; ++p) a.q[4 + p] = h11 * zf2[p];

    const int kbase = ks * 32 + hi * 8;   // kappa chunks at kbase and kbase+16
    {  // N-tile 0 (cols 0..15): 2x ds_load_b128
      U16 b;
      b.p[0] = *(const v8h*)(bp0 + kbase);
      b.p[1] = *(const v8h*)(bp0 + kbase + 16);
      acc2[0] = WMMA(a.v, b.v, acc2[0]);
    }
    {  // N-tile 1 (cols 16..21, rest zeroed by vector cndmask)
      const v8h t0 = *(const v8h*)(bp1 + kbase);
      const v8h t1 = *(const v8h*)(bp1 + kbase + 16);
      U16 b;
      b.p[0] = n1ok ? t0 : z8h;
      b.p[1] = n1ok ? t1 : z8h;
      acc2[1] = WMMA(a.v, b.v, acc2[1]);
    }
  }

  // --- stage D (reusing scratch), then sh-mul + branchless atomic scatter --
  lds_fence_wave();  // all reads of h staging complete before overwrite
  float* dg = (float*)scr[wave];  // 16 rows x 24 floats (96B rows, 16B align)
#pragma unroll
  for (int r = 0; r < 8; ++r) {
    dg[(r + hi * 8) * 24 + l16] = acc2[0][r];              // cols 0..15
  }
#pragma unroll
  for (int r = 0; r < 8; ++r) {
    if (n1ok) dg[(r + hi * 8) * 24 + n1] = acc2[1][r];     // cols 16..21
  }
  lds_fence_wave();

  union { float f[24]; v4f v[6]; } dr;                     // vectorized row read
  const float* dgrow = dg + l16 * 24;
#pragma unroll
  for (int qd = 0; qd < 6; ++qd) dr.v[qd] = *(const v4f*)(dgrow + 4 * qd);

  // materialize all 38 message values with compile-time indexing
  float msg[NMSG];
#pragma unroll
  for (int c = 0; c < NMSG; ++c) {
    if (c < 16) {
      msg[c] = dr.f[c];                                    // scalars (sh0 == 1)
    } else if (c < 28) {
      const int u = (c - 16) / 3, m = (c - 16) % 3;        // vectors
      msg[c] = dr.f[16 + u] * sh[1 + m];
    } else {
      const int u = (c - 28) / 5, m = (c - 28) % 5;        // rank-2 tensors
      msg[c] = dr.f[20 + u] * sh[4 + m];
    }
  }
  // each half-wave scatters 19 columns: hi==0 -> 0..18, hi==1 -> 19..37
  float* obase = out + (size_t)de * OUTC + EMB + hi * 19;
#pragma unroll
  for (int t = 0; t < 19; ++t) {
    const float v = hi ? msg[t + 19] : msg[t];             // cndmask
    atomicAdd(&obase[t], v);
  }
}

// ---------------------------------------------------------------------------
extern "C" void kernel_launch(void* const* d_in, const int* in_sizes, int n_in,
                              void* d_out, int out_size, void* d_ws, size_t ws_size,
                              hipStream_t stream) {
  (void)in_sizes; (void)n_in; (void)out_size; (void)d_ws; (void)ws_size;
  const float* x      = (const float*)d_in[0];
  const float* pos    = (const float*)d_in[1];
  const int*   ei     = (const int*)d_in[2];
  const float* W_emb1 = (const float*)d_in[3];
  const float* W_emb2 = (const float*)d_in[4];
  const float* W_tp1  = (const float*)d_in[5];
  const float* W_tp2  = (const float*)d_in[6];
  float* out = (float*)d_out;

  const int ntot = N_NODES * OUTC;
  zero_kernel<<<(ntot + 255) / 256, 256, 0, stream>>>(out, ntot);
  // 32768/16 = 2048 node tiles, 4 waves/block -> 512 blocks
  node_kernel<<<512, 128, 0, stream>>>(x, W_emb1, W_emb2, out);
  // 262144/16 = 16384 edge tiles, 4 waves/block -> 4096 blocks
  edge_kernel<<<4096, 128, 0, stream>>>(pos, ei, W_tp1, W_tp2, out);
}
